// GraphReasoner_20985210208349
// MI455X (gfx1250) — matrix operand order repaired and checked by reading, not compiled, not run on hardware
//
#include <hip/hip_runtime.h>

typedef __attribute__((ext_vector_type(16))) _Float16 v16h;
typedef __attribute__((ext_vector_type(8)))  float    v8f;
typedef _Float16 f16;

#define BATCH   131072
#define NHEADS  8
#define NSTEPS  3
#define MTILE   64
#define TPB     256   // 8 waves of 32
#define NWAVES  8

// ---------------- f16 weight workspace layout: each block is [N][K] row-major ----------------
constexpr int SZ_WQKV = 768 * 256;   // in_w (identity copy)        : ctx @ in_w.T
constexpr int SZ_WOUT = 256 * 256;   // out_w (identity copy)       : o @ out_w.T
constexpr int SZ_WM1  = 256 * 512;   // m_w1^T                      : combined @ m_w1
constexpr int SZ_WM2  = 256 * 256;   // m_w2^T
constexpr int SZ_WRZ  = 512 * 512;   // [g_wih | g_whh] r,z gates, transposed, K=[msg|h]
constexpr int SZ_WIN  = 256 * 256;   // g_wih n-gate ^T
constexpr int SZ_WHN  = 256 * 256;   // g_whh n-gate ^T
constexpr int SZ_WR1  = 256 * 256;   // r_w1^T
constexpr int SZ_WR2  = 256 * 256;   // r_w2^T
constexpr int OFF_WQKV = 0;
constexpr int OFF_WOUT = OFF_WQKV + SZ_WQKV;
constexpr int OFF_WM1  = OFF_WOUT + SZ_WOUT;
constexpr int OFF_WM2  = OFF_WM1 + SZ_WM1;
constexpr int OFF_WRZ  = OFF_WM2 + SZ_WM2;
constexpr int OFF_WIN  = OFF_WRZ + SZ_WRZ;
constexpr int OFF_WHN  = OFF_WIN + SZ_WIN;
constexpr int OFF_WR1  = OFF_WHN + SZ_WHN;
constexpr int OFF_WR2  = OFF_WR1 + SZ_WR1;
constexpr int W_TOTAL  = OFF_WR2 + SZ_WR2;   // 983,040 f16 elems (~1.92 MB) in d_ws

// ---------------- weight conversion / transposition (runs every launch; deterministic) ------
__global__ void prep_weights(const float* __restrict__ in_w,  const float* __restrict__ out_w,
                             const float* __restrict__ m_w1,  const float* __restrict__ m_w2,
                             const float* __restrict__ g_wih, const float* __restrict__ g_whh,
                             const float* __restrict__ r_w1,  const float* __restrict__ r_w2,
                             f16* __restrict__ W) {
  int i = blockIdx.x * blockDim.x + threadIdx.x;
  if (i >= W_TOTAL) return;
  float v;
  if (i < OFF_WOUT) {                       // WQKV[n][k] = in_w[n][k]   (in_w: [768,256])
    v = in_w[i - OFF_WQKV];
  } else if (i < OFF_WM1) {                 // WOUT[n][k] = out_w[n][k]
    v = out_w[i - OFF_WOUT];
  } else if (i < OFF_WM2) {                 // WM1[n][k] = m_w1[k][n]    (m_w1: [512,256])
    int j = i - OFF_WM1; int n = j >> 9, k = j & 511;
    v = m_w1[k * 256 + n];
  } else if (i < OFF_WRZ) {                 // WM2[n][k] = m_w2[k][n]
    int j = i - OFF_WM2; int n = j >> 8, k = j & 255;
    v = m_w2[k * 256 + n];
  } else if (i < OFF_WIN) {                 // WRZ[n][k]: k<256 -> g_wih[k][n], else g_whh[k-256][n]
    int j = i - OFF_WRZ; int n = j >> 9, k = j & 511;
    v = (k < 256) ? g_wih[k * 768 + n] : g_whh[(k - 256) * 768 + n];
  } else if (i < OFF_WHN) {                 // WIN[n][k] = g_wih[k][512+n]
    int j = i - OFF_WIN; int n = j >> 8, k = j & 255;
    v = g_wih[k * 768 + 512 + n];
  } else if (i < OFF_WR1) {                 // WHN[n][k] = g_whh[k][512+n]
    int j = i - OFF_WHN; int n = j >> 8, k = j & 255;
    v = g_whh[k * 768 + 512 + n];
  } else if (i < OFF_WR2) {                 // WR1[n][k] = r_w1[k][n]
    int j = i - OFF_WR1; int n = j >> 8, k = j & 255;
    v = r_w1[k * 256 + n];
  } else {                                  // WR2[n][k] = r_w2[k][n]
    int j = i - OFF_WR2; int n = j >> 8, k = j & 255;
    v = r_w2[k * 256 + n];
  }
  W[i] = (f16)v;
}

// ---------------- wave-tiled GEMM with 2x2 register blocking ---------------------------------
// Y[MTILE x N] = A(lds f16, lda) * Wt[N][K](global f16, row-major [N][K]).
// Each wave owns 32x32 superblocks: 2 A frags + 2 B frags -> 4 WMMAs per k-step,
// halving LDS + L2 fragment traffic per WMMA vs 1x1 tiling.
// A fragment: CDNA5 16-bit A layout (half-wave K interleave).
// B fragment: same half-wave K interleave as A (column = lane).
template <int N, int K, typename Epi>
__device__ __forceinline__ void gemm_tile(const f16* __restrict__ A, int lda,
                                          const f16* __restrict__ Wt, Epi epi) {
  const int wave = threadIdx.x >> 5;
  const int lane = threadIdx.x & 31;
  const int hl   = lane >> 4;
  const int r    = lane & 15;
  constexpr int NG = N / 32;            // 32-wide column groups
  constexpr int GROUPS = (MTILE / 32) * NG;   // 2 row-groups x NG (all divisible by 8)
  for (int gidx = wave; gidx < GROUPS; gidx += NWAVES) {
    const int mg = gidx / NG;
    const int ng = gidx % NG;
    const f16* arow0 = A  + (mg * 32 + r) * lda + hl * 8;
    const f16* arow1 = arow0 + 16 * lda;
    const f16* brow0 = Wt + (ng * 32 + r) * K + hl * 8;
    const f16* brow1 = brow0 + 16 * K;
    v8f acc00 = {}, acc01 = {}, acc10 = {}, acc11 = {};
    for (int kb = 0; kb < K / 32; ++kb) {
      v16h a0, a1, b0, b1;
      ((float4*)&a0)[0] = *(const float4*)(arow0 + kb * 32);        // K = kb*32 + hl*8 .. +7
      ((float4*)&a0)[1] = *(const float4*)(arow0 + kb * 32 + 16);   // K = +16 .. +23
      ((float4*)&a1)[0] = *(const float4*)(arow1 + kb * 32);
      ((float4*)&a1)[1] = *(const float4*)(arow1 + kb * 32 + 16);
      ((float4*)&b0)[0] = *(const float4*)(brow0 + kb * 32);        // K = kb*32 + hl*8 .. +7
      ((float4*)&b0)[1] = *(const float4*)(brow0 + kb * 32 + 16);   // K = +16 .. +23
      ((float4*)&b1)[0] = *(const float4*)(brow1 + kb * 32);
      ((float4*)&b1)[1] = *(const float4*)(brow1 + kb * 32 + 16);
      acc00 = __builtin_amdgcn_wmma_f32_16x16x32_f16(false, a0, false, b0, (short)0, acc00, false, false);
      acc01 = __builtin_amdgcn_wmma_f32_16x16x32_f16(false, a0, false, b1, (short)0, acc01, false, false);
      acc10 = __builtin_amdgcn_wmma_f32_16x16x32_f16(false, a1, false, b0, (short)0, acc10, false, false);
      acc11 = __builtin_amdgcn_wmma_f32_16x16x32_f16(false, a1, false, b1, (short)0, acc11, false, false);
    }
    const int mb = mg * 32 + hl * 8;     // C layout: vgpr j -> m = mb + j (lanes 0-15) / mb+8.. via hl
    const int n0 = ng * 32 + r;
#pragma unroll
    for (int j = 0; j < 8; ++j) {
      epi(mb + j,      n0,      acc00[j]);
      epi(mb + j,      n0 + 16, acc01[j]);
      epi(mb + 16 + j, n0,      acc10[j]);
      epi(mb + 16 + j, n0 + 16, acc11[j]);
    }
  }
}

// ---------------- LayerNorm(256) + ReLU over an LDS f32 [MTILE x 256] buffer ----------------
// 4 lanes per row, quad reduction via shfl_xor (wave32).
__device__ __forceinline__ void layernorm_relu(const float* __restrict__ raw,
                                               const float* __restrict__ g,
                                               const float* __restrict__ be,
                                               f16* __restrict__ dst) {
  const int t = threadIdx.x;
  const int m = t >> 2, q = t & 3;
  const float* x = raw + m * 256 + q * 64;
  float s = 0.0f;
  for (int j = 0; j < 64; ++j) s += x[j];
  s += __shfl_xor(s, 1); s += __shfl_xor(s, 2);
  const float mu = s * (1.0f / 256.0f);
  float v = 0.0f;
  for (int j = 0; j < 64; ++j) { float d = x[j] - mu; v += d * d; }
  v += __shfl_xor(v, 1); v += __shfl_xor(v, 2);
  const float rstd = rsqrtf(v * (1.0f / 256.0f) + 1e-5f);
  for (int j = 0; j < 64; ++j) {
    const int n = q * 64 + j;
    float y = (x[j] - mu) * rstd * g[n] + be[n];
    dst[m * 256 + n] = (f16)(y > 0.0f ? y : 0.0f);
  }
}

// ---------------- main fused kernel: one 64-row tile per workgroup, h resident in LDS -------
__global__ __launch_bounds__(TPB, 1)
void reasoner_kernel(const float* __restrict__ wm, const float* __restrict__ schema,
                     const float* __restrict__ in_b, const float* __restrict__ out_b,
                     const float* __restrict__ m_b1, const float* __restrict__ m_g,
                     const float* __restrict__ m_be, const float* __restrict__ m_b2,
                     const float* __restrict__ g_bih, const float* __restrict__ g_bhh,
                     const float* __restrict__ r_b1, const float* __restrict__ r_g,
                     const float* __restrict__ r_be, const float* __restrict__ r_b2,
                     const f16* __restrict__ W, float* __restrict__ out) {
  extern __shared__ char smem[];
  f16*  shH  = (f16*)smem;                    // [64x256] persistent hidden state (f16)
  f16*  QKVS = (f16*)(smem + 32768);          // [64x768] schema qkv (computed once)
  char* AR   = smem + 32768 + 98304;          // 160 KB phase-shared arena

  const int row0 = blockIdx.x * MTILE;
  const int tid  = threadIdx.x;

  // ---- load wm -> shH, schema -> arena (both f16) ----
  f16* SCH = (f16*)AR;
  for (int i = tid; i < MTILE * 64; i += TPB) {
    int m = i >> 6, c = (i & 63) * 4;
    float4 a = *(const float4*)(wm     + (size_t)(row0 + m) * 256 + c);
    float4 b = *(const float4*)(schema + (size_t)(row0 + m) * 256 + c);
    f16* hd = shH + m * 256 + c;  f16* sd = SCH + m * 256 + c;
    hd[0]=(f16)a.x; hd[1]=(f16)a.y; hd[2]=(f16)a.z; hd[3]=(f16)a.w;
    sd[0]=(f16)b.x; sd[1]=(f16)b.y; sd[2]=(f16)b.z; sd[3]=(f16)b.w;
  }
  __syncthreads();

  // ---- schema qkv: computed once, reused for all 3 steps ----
  gemm_tile<768, 256>(SCH, 256, W + OFF_WQKV, [&](int m, int n, float v) {
    QKVS[m * 768 + n] = (f16)(v + in_b[n]);
  });
  __syncthreads();

  for (int step = 0; step < NSTEPS; ++step) {
    // Phase A: qkv of h  -> arena[0,96K)
    f16* QKVH = (f16*)AR;
    gemm_tile<768, 256>(shH, 256, W + OFF_WQKV, [&](int m, int n, float v) {
      QKVH[m * 768 + n] = (f16)(v + in_b[n]);
    });
    __syncthreads();

    // Phase B: length-2 attention per (row, head) -> OCAT [64x512] at arena+96K
    f16* OCAT = (f16*)(AR + 98304);
    const float scale = 0.17677669529663687f;   // 1/sqrt(32)
    for (int p = tid; p < MTILE * NHEADS; p += TPB) {
      int m = p >> 3, hd = p & 7;
      const f16* qh = QKVH + m * 768 + hd * 32;
      const f16* kh = qh + 256;  const f16* vh = qh + 512;
      const f16* qs = QKVS + m * 768 + hd * 32;
      const f16* ks = qs + 256;  const f16* vs = qs + 512;
      float s00 = 0, s01 = 0, s10 = 0, s11 = 0;
      for (int d = 0; d < 32; ++d) {
        float qhd = (float)qh[d], khd = (float)kh[d];
        float qsd = (float)qs[d], ksd = (float)ks[d];
        s00 += qhd * khd; s01 += qhd * ksd; s10 += qsd * khd; s11 += qsd * ksd;
      }
      s00 *= scale; s01 *= scale; s10 *= scale; s11 *= scale;
      float mx0 = fmaxf(s00, s01), mx1 = fmaxf(s10, s11);
      float e00 = __expf(s00 - mx0), e01 = __expf(s01 - mx0);
      float e10 = __expf(s10 - mx1), e11 = __expf(s11 - mx1);
      float i0 = 1.0f / (e00 + e01), i1 = 1.0f / (e10 + e11);
      float a00 = e00 * i0, a01 = e01 * i0, a10 = e10 * i1, a11 = e11 * i1;
      for (int d = 0; d < 32; ++d) {
        float vhd = (float)vh[d], vsd = (float)vs[d];
        OCAT[m * 512 + hd * 32 + d]       = (f16)(a00 * vhd + a01 * vsd);
        OCAT[m * 512 + 256 + hd * 32 + d] = (f16)(a10 * vhd + a11 * vsd);
      }
    }
    __syncthreads();

    // Phase B2: out-proj per token -> combined [64x512] at arena[0,64K)
    f16* CMB = (f16*)AR;
    gemm_tile<256, 256>(OCAT, 512, W + OFF_WOUT, [&](int m, int n, float v) {
      CMB[m * 512 + n] = (f16)(v + out_b[n]);
    });
    gemm_tile<256, 256>(OCAT + 256, 512, W + OFF_WOUT, [&](int m, int n, float v) {
      CMB[m * 512 + 256 + n] = (f16)(v + out_b[n]);
    });
    __syncthreads();

    // Phase C: msg_net first linear (K=512) -> f32 raw, then LN+ReLU -> MSGACT f16
    float* M1RAW = (float*)(AR + 65536);
    gemm_tile<256, 512>(CMB, 512, W + OFF_WM1, [&](int m, int n, float v) {
      M1RAW[m * 256 + n] = v + m_b1[n];
    });
    __syncthreads();
    f16* MSGACT = (f16*)(AR + 131072);
    layernorm_relu(M1RAW, m_g, m_be, MSGACT);
    __syncthreads();

    // Phase D: msg = MSGACT @ m_w2 + b  -> XCAT[:,0:256]; copy h -> XCAT[:,256:512]
    f16* XCAT = (f16*)AR;  // [msg | h]  64x512
    for (int i = tid; i < MTILE * 256; i += TPB)
      XCAT[(i >> 8) * 512 + 256 + (i & 255)] = shH[i];
    gemm_tile<256, 256>(MSGACT, 256, W + OFF_WM2, [&](int m, int n, float v) {
      XCAT[m * 512 + n] = (f16)(v + m_b2[n]);
    });
    __syncthreads();

    // Phase E1: fused r,z gates: sigmoid([msg|h] @ Wrz + bih + bhh)
    f16* Rg = (f16*)(AR + 65536);
    f16* Zg = (f16*)(AR + 98304);
    gemm_tile<512, 512>(XCAT, 512, W + OFF_WRZ, [&](int m, int n, float v) {
      float s = 1.0f / (1.0f + __expf(-(v + g_bih[n] + g_bhh[n])));
      if (n < 256) Rg[m * 256 + n] = (f16)s;
      else         Zg[m * 256 + (n - 256)] = (f16)s;
    });
    __syncthreads();

    // Phase E2: hn = h @ Whh_n + bhh_n
    f16* HN = (f16*)(AR + 131072);
    gemm_tile<256, 256>(shH, 256, W + OFF_WHN, [&](int m, int n, float v) {
      HN[m * 256 + n] = (f16)(v + g_bhh[512 + n]);
    });
    __syncthreads();

    // Phase E3: gin = msg @ Wih_n + bih_n; n = tanh(gin + r*hn); h = (1-z)*n + z*h
    gemm_tile<256, 256>(XCAT, 512, W + OFF_WIN, [&](int m, int n, float v) {
      float gin = v + g_bih[512 + n];
      float rr  = (float)Rg[m * 256 + n];
      float hn  = (float)HN[m * 256 + n];
      float nn  = tanhf(gin + rr * hn);
      float zz  = (float)Zg[m * 256 + n];
      float ho  = (float)shH[m * 256 + n];
      shH[m * 256 + n] = (f16)((1.0f - zz) * nn + zz * ho);
    });
    __syncthreads();
  }

  // ---- final head: relu(LN(h @ r_w1 + b)) @ r_w2 + b -> global f32 ----
  float* FRAW = (float*)AR;
  gemm_tile<256, 256>(shH, 256, W + OFF_WR1, [&](int m, int n, float v) {
    FRAW[m * 256 + n] = v + r_b1[n];
  });
  __syncthreads();
  f16* FACT = (f16*)(AR + 65536);
  layernorm_relu(FRAW, r_g, r_be, FACT);
  __syncthreads();
  gemm_tile<256, 256>(FACT, 256, W + OFF_WR2, [&](int m, int n, float v) {
    out[(size_t)(row0 + m) * 256 + n] = v + r_b2[n];
  });
}

// ---------------- host launch ----------------
extern "C" void kernel_launch(void* const* d_in, const int* in_sizes, int n_in,
                              void* d_out, int out_size, void* d_ws, size_t ws_size,
                              hipStream_t stream) {
  (void)in_sizes; (void)n_in; (void)out_size; (void)ws_size;
  const float* wm     = (const float*)d_in[0];
  const float* schema = (const float*)d_in[1];
  const float* in_w   = (const float*)d_in[2];
  const float* in_b   = (const float*)d_in[3];
  const float* out_w  = (const float*)d_in[4];
  const float* out_b  = (const float*)d_in[5];
  const float* m_w1   = (const float*)d_in[6];
  const float* m_b1   = (const float*)d_in[7];
  const float* m_g    = (const float*)d_in[8];
  const float* m_be   = (const float*)d_in[9];
  const float* m_w2   = (const float*)d_in[10];
  const float* m_b2   = (const float*)d_in[11];
  const float* g_wih  = (const float*)d_in[12];
  const float* g_bih  = (const float*)d_in[13];
  const float* g_whh  = (const float*)d_in[14];
  const float* g_bhh  = (const float*)d_in[15];
  const float* r_w1   = (const float*)d_in[16];
  const float* r_b1   = (const float*)d_in[17];
  const float* r_g    = (const float*)d_in[18];
  const float* r_be   = (const float*)d_in[19];
  const float* r_w2   = (const float*)d_in[20];
  const float* r_b2   = (const float*)d_in[21];
  f16*   W   = (f16*)d_ws;
  float* out = (float*)d_out;

  prep_weights<<<(W_TOTAL + 255) / 256, 256, 0, stream>>>(
      in_w, out_w, m_w1, m_w2, g_wih, g_whh, r_w1, r_w2, W);

  constexpr size_t SMEM = 32768 + 98304 + 163840;   // 288 KB per workgroup (<= 320 KB/WGP)
  hipFuncSetAttribute((const void*)reasoner_kernel,
                      hipFuncAttributeMaxDynamicSharedMemorySize, (int)SMEM);
  reasoner_kernel<<<BATCH / MTILE, TPB, SMEM, stream>>>(
      wm, schema, in_b, out_b, m_b1, m_g, m_be, m_b2,
      g_bih, g_bhh, r_b1, r_g, r_be, r_b2, W, out);
}
